// RingConv2dFused_71459665870986
// MI455X (gfx1250) — matrix operand order repaired and checked by reading, not compiled, not run on hardware
//
#include <hip/hip_runtime.h>
#include <hip/hip_bf16.h>

typedef __attribute__((ext_vector_type(16))) _Float16 v16h;
typedef __attribute__((ext_vector_type(8)))  float    v8f;

// Problem constants (from reference): x (4,32,64,64) f32, w (64,32,3,3) f32
// out (4,64,64,64) f32.  P = C*K*K = 288, reduction K = 2P = 576 = 18*32.
#define NB    4
#define NC    32
#define NH    64
#define NW    64
#define NO    64
#define NP    288          // C*K*K
#define HP    66           // padded H/W
#define PAD_ELEMS (NB*NC*HP*HP)        // 557568 halfs per image
#define APACK_ELEMS (4*18*32*16)       // 36864 halfs per packed A matrix

// ---------------------------------------------------------------------------
// Prep 1: padded cos(x)/sin(x) images in f16.  Pad value: cos=1, sin=0
// (a padded zero angle contributes cos(0-w)=cos w, sin(0-w)=-sin w).
// ---------------------------------------------------------------------------
__global__ void ring_prep_img(const float* __restrict__ x,
                              _Float16* __restrict__ cospad,
                              _Float16* __restrict__ sinpad) {
  int i = blockIdx.x * blockDim.x + threadIdx.x;
  if (i >= PAD_ELEMS) return;
  int w66 = i % HP;
  int t   = i / HP;
  int h66 = t % HP;
  int bc  = t / HP;                       // b*32 + c
  float cv = 1.0f, sv = 0.0f;
  if (w66 >= 1 && w66 <= NW && h66 >= 1 && h66 <= NH) {
    float v = x[(bc * NH + (h66 - 1)) * NW + (w66 - 1)];
    sincosf(v, &sv, &cv);
  }
  cospad[i] = (_Float16)cv;
  sinpad[i] = (_Float16)sv;
}

// ---------------------------------------------------------------------------
// Prep 2: pack weights into the CDNA5 16-bit A-matrix (16x32) lane layout.
// A row M = lane%16.  Element j of the per-lane v16h maps to
//   K = 2*(j/2) + (j/2>=4 ? 8:0) + (lane>=16 ? 8:0) + (j&1)     (ISA 7.12.2)
// Global reduction index kg = kit*32 + K.
//   kg <  288 : pairs with cos(x) rows -> Ax = cos(w), Ay = -sin(w)
//   kg >= 288 : pairs with sin(x) rows -> Ax = sin(w), Ay =  cos(w)
// giving dir_x = CwCx + SwSx, dir_y = CwSx - SwCx.
// Storage: AxP[((mtile*18 + kit)*32 + lane)*16 + j]
// ---------------------------------------------------------------------------
__global__ void ring_prep_w(const float* __restrict__ w,
                            _Float16* __restrict__ AxP,
                            _Float16* __restrict__ AyP) {
  int i = blockIdx.x * blockDim.x + threadIdx.x;
  if (i >= APACK_ELEMS) return;
  int j    = i & 15;
  int lane = (i >> 4) & 31;
  int kit  = (i >> 9) % 18;
  int mt   = i / 9216;
  int v    = j >> 1;
  int K    = 2 * v + (v >= 4 ? 8 : 0) + (lane >= 16 ? 8 : 0) + (j & 1);
  int kg   = kit * 32 + K;
  int m    = mt * 16 + (lane & 15);
  float ax, ay;
  if (kg < NP) {
    float wv = w[m * NP + kg];
    float c, s; sincosf(wv, &s, &c);
    ax = c; ay = -s;
  } else {
    float wv = w[m * NP + kg - NP];
    float c, s; sincosf(wv, &s, &c);
    ax = s; ay = c;
  }
  AxP[i] = (_Float16)ax;
  AyP[i] = (_Float16)ay;
}

// ---------------------------------------------------------------------------
// Main: implicit-GEMM ring conv.  grid = 1024 N-tiles (16 consecutive output
// columns of one image row), block = 128 threads = 4 waves = 4 M-tiles (all 64
// output channels).
//   Phase 1: stage 32ch x 3row x 18col cos/sin f16 window -> LDS (coalesced).
//   Phase 2: cooperative im2col swizzle ONCE per block into WMMA B lane
//            order: Bpanel[k2][lane][16] (B rows: lanes 0-15 -> K 0..7,16..23,
//            lanes 16-31 -> K 8..15,24..31; column N = lane%16).
//   Phase 3: 9 k-steps, each wave: 2x contiguous 32B LDS B loads + 4x
//            coalesced 32B packed-A global loads + 4x v_wmma_f32_16x16x32_f16.
// ---------------------------------------------------------------------------
__global__ __launch_bounds__(128, 2) void ring_main(
    const _Float16* __restrict__ cospad, const _Float16* __restrict__ sinpad,
    const _Float16* __restrict__ AxP,    const _Float16* __restrict__ AyP,
    float* __restrict__ out) {
  __shared__ _Float16 cimg[NC * 3 * 18];   // 1728 halfs
  __shared__ _Float16 simg[NC * 3 * 18];
  __shared__ _Float16 BpanC[9 * 32 * 16];  // 4608 halfs, 32B per (k2,lane) row
  __shared__ _Float16 BpanS[9 * 32 * 16];

  const int tid  = threadIdx.x;
  const int lane = tid & 31;
  const int wave = tid >> 5;               // M-tile 0..3
  const int n0   = blockIdx.x * 16;
  const int b    = n0 >> 12;               // / 4096
  const int loc  = n0 & 4095;
  const int oy   = loc >> 6;
  const int ox0  = loc & 63;               // multiple of 16

  // L2 warm-up for this wave's packed-A streams (first k-step slice only).
  __builtin_prefetch(AxP + (wave * 18 * 32 + lane) * 16, 0, 1);
  __builtin_prefetch(AyP + (wave * 18 * 32 + lane) * 16, 0, 1);

  // Phase 1: coalesced stage of the padded-image window (all 18 k-steps).
  for (int i = tid; i < NC * 3 * 18; i += 128) {
    int col = i % 18;
    int t   = i / 18;
    int r   = t % 3;
    int c   = t / 3;
    int g   = ((b * NC + c) * HP + (oy + r)) * HP + (ox0 + col);
    cimg[i] = cospad[g];
    simg[i] = sinpad[g];
  }
  __syncthreads();

  // Phase 2: build WMMA-ordered B panels once per block (no per-wave redo).
  for (int row = tid; row < 9 * 32; row += 128) {
    int k2  = row >> 5;
    int lt  = row & 31;
    int nct = lt & 15;
    int kht = (lt >> 4) << 4;
    v16h vc, vs;
#pragma unroll
    for (int j = 0; j < 16; ++j) {
      int p   = k2 * 32 + (kht >> 1) + j + ((j >> 3) << 3);  // patch element 0..287
      int c   = p / 9;
      int rs  = p - c * 9;
      int r   = rs / 3;
      int s   = rs - r * 3;
      int idx = (c * 3 + r) * 18 + s + nct;
      vc[j] = cimg[idx];
      vs[j] = simg[idx];
    }
    *(v16h*)(BpanC + row * 16) = vc;
    *(v16h*)(BpanS + row * 16) = vs;
  }
  __syncthreads();

  // Phase 3: GEMM main loop.
  v8f accX = {};
  v8f accY = {};
#pragma unroll 3
  for (int k2 = 0; k2 < 9; ++k2) {
    const v16h bc = *(const v16h*)(BpanC + (k2 * 32 + lane) * 16);
    const v16h bs = *(const v16h*)(BpanS + (k2 * 32 + lane) * 16);
    const v16h axc = *(const v16h*)(AxP + ((wave * 18 + k2    ) * 32 + lane) * 16);
    const v16h axs = *(const v16h*)(AxP + ((wave * 18 + k2 + 9) * 32 + lane) * 16);
    const v16h ayc = *(const v16h*)(AyP + ((wave * 18 + k2    ) * 32 + lane) * 16);
    const v16h ays = *(const v16h*)(AyP + ((wave * 18 + k2 + 9) * 32 + lane) * 16);

    accX = __builtin_amdgcn_wmma_f32_16x16x32_f16(false, axc, false, bc,
                                                  (short)0, accX, false, false);
    accX = __builtin_amdgcn_wmma_f32_16x16x32_f16(false, axs, false, bs,
                                                  (short)0, accX, false, false);
    accY = __builtin_amdgcn_wmma_f32_16x16x32_f16(false, ayc, false, bc,
                                                  (short)0, accY, false, false);
    accY = __builtin_amdgcn_wmma_f32_16x16x32_f16(false, ays, false, bs,
                                                  (short)0, accY, false, false);
  }

  // Epilogue: C/D layout -> VGPR i holds row M = i + (lane>=16 ? 8 : 0),
  // column = lane%16.  Channel o = wave*16 + M.
  const int ncol  = lane & 15;
  const int obase = (b * NO + wave * 16 + ((lane >> 4) << 3)) * 4096
                  + oy * 64 + ox0 + ncol;
#pragma unroll
  for (int i = 0; i < 8; ++i) {
    out[obase + i * 4096] = atan2f(accY[i], accX[i]);
  }
}

// ---------------------------------------------------------------------------
extern "C" void kernel_launch(void* const* d_in, const int* in_sizes, int n_in,
                              void* d_out, int out_size, void* d_ws, size_t ws_size,
                              hipStream_t stream) {
  const float* x = (const float*)d_in[0];   // (4,32,64,64)
  const float* w = (const float*)d_in[1];   // (64,32,3,3)
  float* out = (float*)d_out;               // (4,64,64,64)

  char* ws = (char*)d_ws;
  _Float16* cospad = (_Float16*)(ws);                          // 1,115,136 B
  _Float16* sinpad = (_Float16*)(ws + 1115136);                // 1,115,136 B
  _Float16* AxP    = (_Float16*)(ws + 2230272);                //    73,728 B
  _Float16* AyP    = (_Float16*)(ws + 2304000);                //    73,728 B

  ring_prep_img<<<(PAD_ELEMS + 255) / 256, 256, 0, stream>>>(x, cospad, sinpad);
  ring_prep_w<<<(APACK_ELEMS + 255) / 256, 256, 0, stream>>>(w, AxP, AyP);
  ring_main<<<1024, 128, 0, stream>>>(cospad, sinpad, AxP, AyP, out);
}
